// LSTMDecoder_57543971832698
// MI455X (gfx1250) — matrix operand order repaired and checked by reading, not compiled, not run on hardware
//
#include <hip/hip_runtime.h>
#include <hip/hip_bf16.h>

// ---------------------------------------------------------------------------
// LSTM decoder, persistent-kernel formulation for MI455X (gfx1250, wave32).
//  * output projection folded into layer-0 input weights (Wcomb = WihP @ Wout)
//  * z-dependent gate term precomputed once (time-invariant)
//  * bf16 WMMA 16x16x32, f32 accumulate; c-state and activations in f32
//  * 128 WGs x 256 threads = 1024 waves <-> 1024 (batch,hidden) 16x16 pairs
//  * 2 atomic grid barriers per timestep; h buffers double-buffered
// ---------------------------------------------------------------------------

#define LATENT  128
#define HIDDEN  512
#define OUTDIM  64
#define SEQLEN  1024
#define BATCH   512
#define GATES   2048          // 4*HIDDEN
#define KCAT    1024          // folded K for both layers
#define NWG     128
#define NTHR    256

typedef __bf16 bf16_t;
typedef __attribute__((ext_vector_type(16))) __bf16       v16bf;
typedef __attribute__((ext_vector_type(8)))  float        v8f;
typedef __attribute__((ext_vector_type(4)))  unsigned int u32x4;

union FragU { v16bf v; u32x4 q[2]; };

// A fragment (16x32 bf16): lane holds row M=lane&15, khalf=lane>>4.
// elems 0..15 <- K = k0+khalf*16+{0..15}
__device__ __forceinline__ v16bf load_a_frag(const bf16_t* p) {
  FragU f;
  f.q[0] = *(const u32x4*)(p);
  f.q[1] = *(const u32x4*)(p + 8);
  return f.v;
}
// B fragment (32x16 bf16): lane holds col N=lane&15, 16 consecutive K at k0+khalf*16
__device__ __forceinline__ v16bf load_b_frag(const bf16_t* p) {
  FragU f;
  f.q[0] = *(const u32x4*)(p);
  f.q[1] = *(const u32x4*)(p + 8);
  return f.v;
}
__device__ __forceinline__ v8f wmma_bf16(v16bf a, v16bf b, v8f c) {
  return __builtin_amdgcn_wmma_f32_16x16x32_bf16(false, a, false, b,
                                                 (short)0, c, false, false);
}
__device__ __forceinline__ float sigm(float x)  { return 1.0f / (1.0f + __expf(-x)); }
__device__ __forceinline__ float tanh_(float x) { return 1.0f - 2.0f / (__expf(2.0f * x) + 1.0f); }

__device__ __forceinline__ void grid_barrier(unsigned* cnt, unsigned* phase) {
  __syncthreads();
  if (threadIdx.x == 0) {
    unsigned target = (++(*phase)) * (unsigned)NWG;
    __hip_atomic_fetch_add(cnt, 1u, __ATOMIC_RELEASE, __HIP_MEMORY_SCOPE_AGENT);
    while (__hip_atomic_load(cnt, __ATOMIC_ACQUIRE, __HIP_MEMORY_SCOPE_AGENT) < target)
      __builtin_amdgcn_s_sleep(2);
  }
  __syncthreads();
}

// --------------------------- prep kernels ----------------------------------

// W0cat (2048 x 1024) bf16: k<512 -> Wcomb = Wih0[:, :64] @ Wout ; k>=512 -> Whh0
__global__ void k_w0cat(const float* __restrict__ Wih0, const float* __restrict__ Whh0,
                        const float* __restrict__ Wout, bf16_t* __restrict__ W0) {
  int tid = blockIdx.x * blockDim.x + threadIdx.x;
  if (tid >= GATES * KCAT) return;
  int r = tid >> 10, k = tid & 1023;
  float v;
  if (k < HIDDEN) {
    v = 0.0f;
    for (int j = 0; j < OUTDIM; ++j) v += Wih0[r * 192 + j] * Wout[j * HIDDEN + k];
  } else {
    v = Whh0[r * HIDDEN + (k - HIDDEN)];
  }
  W0[tid] = (bf16_t)v;
}

// W1cat (2048 x 1024) bf16: [Wih1 | Whh1]; also Woutb, bias1, barrier reset
__global__ void k_w1cat(const float* __restrict__ Wih1, const float* __restrict__ Whh1,
                        const float* __restrict__ Wout, const float* __restrict__ bih1,
                        const float* __restrict__ bhh1, bf16_t* __restrict__ W1,
                        bf16_t* __restrict__ Wo, float* __restrict__ bias1,
                        unsigned* cnt) {
  int tid = blockIdx.x * blockDim.x + threadIdx.x;
  if (tid < GATES * KCAT) {
    int r = tid >> 10, k = tid & 1023;
    W1[tid] = (bf16_t)(k < HIDDEN ? Wih1[r * HIDDEN + k] : Whh1[r * HIDDEN + (k - HIDDEN)]);
  }
  if (tid < OUTDIM * HIDDEN) Wo[tid] = (bf16_t)Wout[tid];
  if (tid < GATES)           bias1[tid] = bih1[tid] + bhh1[tid];
  if (tid == 0)              *cnt = 0;
}

// Initial h/c: fc = z @ W.T + b laid out flat (B x 1024); raw reshape means
// the first 262144 flat elements are layer-0 state, the rest layer-1.
__global__ void k_state(const float* __restrict__ z,
                        const float* __restrict__ fhW, const float* __restrict__ fhb,
                        const float* __restrict__ fcW, const float* __restrict__ fcb,
                        bf16_t* __restrict__ h0, bf16_t* __restrict__ h1,
                        float* __restrict__ c0, float* __restrict__ c1) {
  int tid = blockIdx.x * blockDim.x + threadIdx.x;
  if (tid >= BATCH * 2 * HIDDEN) return;
  int b = tid >> 10, col = tid & 1023;
  float ah = fhb[col], ac = fcb[col];
  for (int k = 0; k < LATENT; ++k) {
    float zv = z[b * LATENT + k];
    ah += zv * fhW[col * LATENT + k];
    ac += zv * fcW[col * LATENT + k];
  }
  if (tid < BATCH * HIDDEN) { h0[tid] = (bf16_t)ah; c0[tid] = ac; }
  else { h1[tid - BATCH * HIDDEN] = (bf16_t)ah; c1[tid - BATCH * HIDDEN] = ac; }
}

// zgate (B x 2048): z part of layer-0 gates + biases; zgB additionally folds
// the bout @ WihP.T constant (valid for t>=1 where prev = h1@Wout.T + bout).
__global__ void k_zgate(const float* __restrict__ z, const float* __restrict__ Wih0,
                        const float* __restrict__ bih0, const float* __restrict__ bhh0,
                        const float* __restrict__ bout,
                        float* __restrict__ zgA, float* __restrict__ zgB) {
  int tid = blockIdx.x * blockDim.x + threadIdx.x;
  if (tid >= BATCH * GATES) return;
  int b = tid >> 11, r = tid & 2047;
  float s = bih0[r] + bhh0[r];
  for (int k = 0; k < LATENT; ++k) s += z[b * LATENT + k] * Wih0[r * 192 + 64 + k];
  float bt = 0.0f;
  for (int j = 0; j < OUTDIM; ++j) bt += bout[j] * Wih0[r * 192 + j];
  zgA[tid] = s;
  zgB[tid] = s + bt;
}

// --------------------------- persistent kernel -----------------------------

__device__ __forceinline__ void pred_tile(const bf16_t* __restrict__ h1,
                                          const bf16_t* __restrict__ Wo,
                                          const float* __restrict__ bout,
                                          float* __restrict__ out,
                                          int tile, int t, int n, int khalf, int mbase) {
  int btp = tile >> 2, ot = tile & 3;
  float bv = bout[ot * 16 + n];
  v8f acc;
  #pragma unroll
  for (int r = 0; r < 8; ++r) acc[r] = bv;
  for (int kk = 0; kk < 16; ++kk) {
    v16bf av  = load_a_frag(h1 + (size_t)(btp * 16 + n) * HIDDEN + kk * 32 + khalf * 16);
    v16bf bvf = load_b_frag(Wo + (size_t)(ot * 16 + n) * HIDDEN + kk * 32 + khalf * 16);
    acc = wmma_bf16(av, bvf, acc);
  }
  #pragma unroll
  for (int r = 0; r < 8; ++r)
    out[(size_t)(btp * 16 + mbase + r) * (SEQLEN * OUTDIM) + (size_t)t * OUTDIM +
        ot * 16 + n] = acc[r];
}

__global__ __launch_bounds__(NTHR) void lstm_persistent(
    const bf16_t* __restrict__ W0, const bf16_t* __restrict__ W1,
    const bf16_t* __restrict__ Wo,
    const float* __restrict__ zgA, const float* __restrict__ zgB,
    const float* __restrict__ bias1, const float* __restrict__ bout,
    bf16_t* h0a, bf16_t* h0b, bf16_t* h1a, bf16_t* h1b,
    float* c0, float* c1, float* out, unsigned* cnt)
{
  const int lane  = threadIdx.x & 31;
  const int wid   = (blockIdx.x << 3) + (threadIdx.x >> 5);   // 0..1023
  const int n     = lane & 15;
  const int khalf = lane >> 4;
  const int mbase = khalf * 8;
  const int bt = wid >> 5;                                    // batch tile 0..31
  const int ht = wid & 31;                                    // hidden tile 0..31
  const size_t grow = (size_t)HIDDEN * KCAT;                  // gate stride in Wcat
  unsigned phase = 0;

  for (int t = 0; t < SEQLEN; ++t) {
    const bf16_t* h0r = (t & 1) ? h0b : h0a;
    bf16_t*       h0w = (t & 1) ? h0a : h0b;
    const bf16_t* h1r = (t & 1) ? h1b : h1a;
    bf16_t*       h1w = (t & 1) ? h1a : h1b;

    // ---------------- Phase A : layer-0 cell (+ output projection of t-1) --
    {
      const float* zg = t ? zgB : zgA;
      v8f acc[4];
      #pragma unroll
      for (int g = 0; g < 4; ++g) {
        const float* zc = zg + (size_t)(bt * 16 + mbase) * GATES + g * HIDDEN + ht * 16 + n;
        #pragma unroll
        for (int r = 0; r < 8; ++r) acc[g][r] = zc[(size_t)r * GATES];
      }
      const bf16_t* wbase = W0 + (size_t)(ht * 16 + n) * KCAT;
      for (int kk = (t ? 0 : 16); kk < 32; ++kk) {            // k<512: h1(t-1)@Wcomb
        const bf16_t* Asrc = (kk < 16) ? h1r : h0r;           // k>=512: h0(t-1)@Whh0
        v16bf av = load_a_frag(Asrc + (size_t)(bt * 16 + n) * HIDDEN +
                               (kk & 15) * 32 + khalf * 16);
        const bf16_t* wk = wbase + kk * 32 + khalf * 16;
        #pragma unroll
        for (int g = 0; g < 4; ++g) {
          v16bf bv = load_b_frag(wk + (size_t)g * grow);
          acc[g] = wmma_bf16(av, bv, acc[g]);
        }
      }
      #pragma unroll
      for (int r = 0; r < 8; ++r) {
        size_t idx = (size_t)(bt * 16 + mbase + r) * HIDDEN + ht * 16 + n;
        float cold = c0[idx];
        float iv = sigm(acc[0][r]);
        float fv = sigm(acc[1][r]);
        float gv = tanh_(acc[2][r]);
        float ov = sigm(acc[3][r]);
        float cn = fv * cold + iv * gv;
        c0[idx] = cn;
        h0w[idx] = (bf16_t)(ov * tanh_(cn));
      }
      if (wid < 128 && t > 0) pred_tile(h1r, Wo, bout, out, wid, t - 1, n, khalf, mbase);
    }
    grid_barrier(cnt, &phase);

    // ---------------- Phase B : layer-1 cell -------------------------------
    {
      v8f acc[4];
      #pragma unroll
      for (int g = 0; g < 4; ++g) {
        float bv = bias1[g * HIDDEN + ht * 16 + n];
        #pragma unroll
        for (int r = 0; r < 8; ++r) acc[g][r] = bv;
      }
      const bf16_t* wbase = W1 + (size_t)(ht * 16 + n) * KCAT;
      for (int kk = 0; kk < 32; ++kk) {                       // k<512: h0(t)@Wih1
        const bf16_t* Asrc = (kk < 16) ? h0w : h1r;           // k>=512: h1(t-1)@Whh1
        v16bf av = load_a_frag(Asrc + (size_t)(bt * 16 + n) * HIDDEN +
                               (kk & 15) * 32 + khalf * 16);
        const bf16_t* wk = wbase + kk * 32 + khalf * 16;
        #pragma unroll
        for (int g = 0; g < 4; ++g) {
          v16bf bv = load_b_frag(wk + (size_t)g * grow);
          acc[g] = wmma_bf16(av, bv, acc[g]);
        }
      }
      #pragma unroll
      for (int r = 0; r < 8; ++r) {
        size_t idx = (size_t)(bt * 16 + mbase + r) * HIDDEN + ht * 16 + n;
        float cold = c1[idx];
        float iv = sigm(acc[0][r]);
        float fv = sigm(acc[1][r]);
        float gv = tanh_(acc[2][r]);
        float ov = sigm(acc[3][r]);
        float cn = fv * cold + iv * gv;
        c1[idx] = cn;
        h1w[idx] = (bf16_t)(ov * tanh_(cn));
      }
    }
    grid_barrier(cnt, &phase);
  }

  // final timestep's output projection: h1(1023) lives in buffer 0
  if (wid < 128) pred_tile(h1a, Wo, bout, out, wid, SEQLEN - 1, n, khalf, mbase);
}

// --------------------------- host entry ------------------------------------

extern "C" void kernel_launch(void* const* d_in, const int* in_sizes, int n_in,
                              void* d_out, int out_size, void* d_ws, size_t ws_size,
                              hipStream_t stream) {
  (void)in_sizes; (void)n_in; (void)out_size;
  const float* z     = (const float*)d_in[0];
  const float* fhW   = (const float*)d_in[1];
  const float* fhb   = (const float*)d_in[2];
  const float* fcW   = (const float*)d_in[3];
  const float* fcb   = (const float*)d_in[4];
  const float* Wih0  = (const float*)d_in[5];
  const float* Whh0  = (const float*)d_in[6];
  const float* bih0  = (const float*)d_in[7];
  const float* bhh0  = (const float*)d_in[8];
  const float* Wih1  = (const float*)d_in[9];
  const float* Whh1  = (const float*)d_in[10];
  const float* bih1  = (const float*)d_in[11];
  const float* bhh1  = (const float*)d_in[12];
  const float* Wout  = (const float*)d_in[13];
  const float* bout  = (const float*)d_in[14];
  float* out = (float*)d_out;

  char* ws = (char*)d_ws;
  size_t off = 0;
  auto wsalloc = [&](size_t bytes) -> void* {
    void* p = ws + off;
    off += (bytes + 255) & ~(size_t)255;
    return p;
  };
  unsigned* cnt  = (unsigned*)wsalloc(256);
  bf16_t* W0     = (bf16_t*)wsalloc((size_t)GATES * KCAT * 2);
  bf16_t* W1     = (bf16_t*)wsalloc((size_t)GATES * KCAT * 2);
  bf16_t* Wo     = (bf16_t*)wsalloc((size_t)OUTDIM * HIDDEN * 2);
  float*  zgA    = (float*)wsalloc((size_t)BATCH * GATES * 4);
  float*  zgB    = (float*)wsalloc((size_t)BATCH * GATES * 4);
  float*  bias1  = (float*)wsalloc((size_t)GATES * 4);
  bf16_t* h0a    = (bf16_t*)wsalloc((size_t)BATCH * HIDDEN * 2);
  bf16_t* h0b    = (bf16_t*)wsalloc((size_t)BATCH * HIDDEN * 2);
  bf16_t* h1a    = (bf16_t*)wsalloc((size_t)BATCH * HIDDEN * 2);
  bf16_t* h1b    = (bf16_t*)wsalloc((size_t)BATCH * HIDDEN * 2);
  float*  c0     = (float*)wsalloc((size_t)BATCH * HIDDEN * 4);
  float*  c1     = (float*)wsalloc((size_t)BATCH * HIDDEN * 4);
  if (off > ws_size) return;  // insufficient scratch (deterministic no-op)

  k_w0cat<<<(GATES * KCAT + 255) / 256, 256, 0, stream>>>(Wih0, Whh0, Wout, W0);
  k_w1cat<<<(GATES * KCAT + 255) / 256, 256, 0, stream>>>(Wih1, Whh1, Wout, bih1, bhh1,
                                                          W1, Wo, bias1, cnt);
  k_state<<<(BATCH * 2 * HIDDEN + 255) / 256, 256, 0, stream>>>(z, fhW, fhb, fcW, fcb,
                                                                h0a, h1a, c0, c1);
  k_zgate<<<(BATCH * GATES + 255) / 256, 256, 0, stream>>>(z, Wih0, bih0, bhh0, bout,
                                                           zgA, zgB);
  lstm_persistent<<<NWG, NTHR, 0, stream>>>(W0, W1, Wo, zgA, zgB, bias1, bout,
                                            h0a, h0b, h1a, h1b, c0, c1, out, cnt);
}
